// LocalWindowAttention_37984690765920
// MI455X (gfx1250) — matrix operand order repaired and checked by reading, not compiled, not run on hardware
//
#include <hip/hip_runtime.h>
#include <hip/hip_bf16.h>

typedef __bf16 bf16;
typedef bf16  v16bf __attribute__((ext_vector_type(16)));
typedef bf16  v8bf  __attribute__((ext_vector_type(8)));
typedef bf16  v4bf  __attribute__((ext_vector_type(4)));
typedef float v8f   __attribute__((ext_vector_type(8)));
typedef float v4f   __attribute__((ext_vector_type(4)));   // for NT vector ld/st

#define B_   8
#define NB_  1024
#define C_   1024
#define T_   32
#define D_   1024
#define M_   (B_ * NB_)   // 8192 rows for all GEMMs

// ---------------------------------------------------------------------------
// Kernel 1: tokens[b,n,c] = mean over T of xb[b,n,c,t]  -> bf16
// xb is a pure 1.07GB stream: non-temporal loads keep it out of L2 so the
// GEMM working set (tokens/weights/QKV, ~150MB) stays resident in 192MB L2.
// ---------------------------------------------------------------------------
__global__ __launch_bounds__(256) void mean_kernel(const v4f* __restrict__ xb,
                                                   bf16* __restrict__ tokb) {
    size_t idx = (size_t)blockIdx.x * 256 + threadIdx.x;   // over B*NB*C = 8M
    const v4f* p = xb + idx * 8;                           // 8 x v4f = 32 floats
    float s = 0.f;
#pragma unroll
    for (int i = 0; i < 8; ++i) {
        v4f v = __builtin_nontemporal_load(p + i);
        s += v[0] + v[1] + v[2] + v[3];
    }
    tokb[idx] = (bf16)(s * (1.0f / 32.0f));
}

// ---------------------------------------------------------------------------
// Kernel 2: Wt[n*K + k] = (bf16) W[k*N + n]   (transpose + convert)
// ---------------------------------------------------------------------------
__global__ __launch_bounds__(256) void wtrans_kernel(const float* __restrict__ W,
                                                     bf16* __restrict__ Wt,
                                                     int K, int N) {
    int idx = blockIdx.x * 256 + threadIdx.x;  // linear over output (n-major)
    if (idx >= K * N) return;
    int n = idx / K;
    int k = idx - n * K;
    Wt[idx] = (bf16)W[(size_t)k * N + n];
}

// ---------------------------------------------------------------------------
// Kernel 3/5: bf16 WMMA GEMM: out(M x N) = A(M x K, row-major bf16)
//                                        * Bt(N x K, row-major bf16)^T (+pos)
// One wave computes a 64x64 tile: 4x4 grid of 16x16x32 WMMA accumulators.
// Wave32 fragments: A/B use contiguous k = 16*(lane/16) + s, not the ISA table order.
// ---------------------------------------------------------------------------
template <bool BF16OUT, bool HAS_POS>
__global__ __launch_bounds__(128)
void gemm64_kernel(const bf16* __restrict__ A, const bf16* __restrict__ Bt,
                   const float* __restrict__ pos,   // (NB x N), used iff HAS_POS
                   float* __restrict__ outf, bf16* __restrict__ outb,
                   int M, int N, int K) {
    const int lane  = threadIdx.x & 31;
    const int wave  = threadIdx.x >> 5;
    const int tile  = blockIdx.x * 4 + wave;
    const int tn_n  = N >> 6;                 // tiles along N
    const int tm    = tile / tn_n;
    const int tn    = tile - tm * tn_n;
    const int row0  = tm << 6;
    const int col0  = tn << 6;
    const int mlane = lane & 15;
    const int sel   = lane >> 4;              // lane half selects K sub-range

    v8f acc[4][4] = {};

    for (int k0 = 0; k0 < K; k0 += 32) {
        v16bf af[4], bfr[4];
        // A fragments: lane holds row (row0+i*16+mlane); 16 consecutive K:
        //   sel=0: K {0..15}
        //   sel=1: K {16..31}
#pragma unroll
        for (int i = 0; i < 4; ++i) {
            const bf16* ap = A + (size_t)(row0 + i * 16 + mlane) * K + k0;
            v8bf p0 = *(const v8bf*)(ap + sel * 16);
            v8bf p1 = *(const v8bf*)(ap + sel * 16 + 8);
#pragma unroll
            for (int t = 0; t < 8; ++t) { af[i][t] = p0[t]; af[i][8 + t] = p1[t]; }
        }
        // B fragments: lane holds column (col0+j*16+mlane); 16 consecutive K
        // starting at k0 + sel*16 (contiguous in N-major Bt -> one 32B load)
#pragma unroll
        for (int j = 0; j < 4; ++j) {
            const bf16* bp = Bt + (size_t)(col0 + j * 16 + mlane) * K + k0 + sel * 16;
            bfr[j] = *(const v16bf*)bp;
        }
#pragma unroll
        for (int i = 0; i < 4; ++i)
#pragma unroll
            for (int j = 0; j < 4; ++j)
                acc[i][j] = __builtin_amdgcn_wmma_f32_16x16x32_bf16(
                    false, af[i], false, bfr[j], (short)0, acc[i][j], false, false);
    }

    // C/D layout: VGPR r -> M = r (lanes 0-15) or r+8 (lanes 16-31), N = lane&15
#pragma unroll
    for (int i = 0; i < 4; ++i)
#pragma unroll
        for (int j = 0; j < 4; ++j) {
            int colg = col0 + j * 16 + mlane;
#pragma unroll
            for (int r = 0; r < 8; ++r) {
                int   rowg = row0 + i * 16 + sel * 8 + r;
                float v    = acc[i][j][r];
                if (HAS_POS) v += pos[(size_t)(rowg & (NB_ - 1)) * N + colg];
                if (BF16OUT) outb[(size_t)rowg * N + colg] = (bf16)v;
                else         outf[(size_t)rowg * N + colg] = v;
            }
        }
}

// ---------------------------------------------------------------------------
// Kernel 4: local window attention (window 9, full-D dot; heads unused in ref)
// One 256-thread block per (b,n); thread owns 4 dims of D=1024.
// ---------------------------------------------------------------------------
#define XSUM(p, m)                                                                 \
    p += __int_as_float(__builtin_amdgcn_ds_swizzle(__float_as_int(p),             \
                                                    (((m) << 10) | 0x1f)))

__global__ __launch_bounds__(256)
void attn_kernel(const v4f* __restrict__ Q, const v4bf* __restrict__ Kb,
                 const v4bf* __restrict__ Vb, v4bf* __restrict__ ctx) {
    const int bn   = blockIdx.x;           // b*1024 + n
    const int n    = bn & (NB_ - 1);
    const int base = bn - n;               // b*1024
    const int tid  = threadIdx.x;

    __shared__ float red[9][8];
    __shared__ float sc[9];

    v4f q = Q[(size_t)bn * 256 + tid];

    int nbr[9];
#pragma unroll
    for (int w = 0; w < 9; ++w) {
        int nb = n + w - 4;
        nb     = nb < 0 ? 0 : (nb > NB_ - 1 ? NB_ - 1 : nb);
        nbr[w] = base + nb;
    }

#pragma unroll
    for (int w = 0; w < 9; ++w) {
        v4bf  kv = Kb[(size_t)nbr[w] * 256 + tid];
        float p  = q[0] * (float)kv[0] + q[1] * (float)kv[1] +
                   q[2] * (float)kv[2] + q[3] * (float)kv[3];
        XSUM(p, 1); XSUM(p, 2); XSUM(p, 4); XSUM(p, 8); XSUM(p, 16);
        if ((tid & 31) == 0) red[w][tid >> 5] = p;
    }
    __syncthreads();
    if (tid < 9) {
        float s = 0.f;
#pragma unroll
        for (int j = 0; j < 8; ++j) s += red[tid][j];
        sc[tid] = s * (1.0f / 32.0f);      // 1/sqrt(D), D=1024
    }
    __syncthreads();

    float mx = sc[0];
#pragma unroll
    for (int w = 1; w < 9; ++w) mx = fmaxf(mx, sc[w]);
    float e[9], sum = 0.f;
#pragma unroll
    for (int w = 0; w < 9; ++w) { e[w] = __expf(sc[w] - mx); sum += e[w]; }
    float inv = 1.0f / sum;

    float c0 = 0.f, c1 = 0.f, c2 = 0.f, c3 = 0.f;
#pragma unroll
    for (int w = 0; w < 9; ++w) {
        float aw = e[w] * inv;
        v4bf  vv = Vb[(size_t)nbr[w] * 256 + tid];
        c0 += aw * (float)vv[0];
        c1 += aw * (float)vv[1];
        c2 += aw * (float)vv[2];
        c3 += aw * (float)vv[3];
    }
    v4bf o;
    o[0] = (bf16)c0; o[1] = (bf16)c1; o[2] = (bf16)c2; o[3] = (bf16)c3;
    ctx[(size_t)bn * 256 + tid] = o;
}

// ---------------------------------------------------------------------------
// Kernel 6: out[b,n,c,:] = xb[b,n,c,:] + out_tok[b,n,c]
// Streaming 2.1GB: non-temporal on both the xb re-read and the output store.
// ---------------------------------------------------------------------------
__global__ __launch_bounds__(256) void addbcast_kernel(const v4f* __restrict__ xb,
                                                       const float* __restrict__ otok,
                                                       v4f* __restrict__ out) {
    size_t idx = (size_t)blockIdx.x * 256 + threadIdx.x;   // over B*NB*C
    float  s   = otok[idx];
#pragma unroll
    for (int i = 0; i < 8; ++i) {
        v4f v = __builtin_nontemporal_load(xb + idx * 8 + i);
        v[0] += s; v[1] += s; v[2] += s; v[3] += s;
        __builtin_nontemporal_store(v, out + idx * 8 + i);
    }
}

// ---------------------------------------------------------------------------
extern "C" void kernel_launch(void* const* d_in, const int* in_sizes, int n_in,
                              void* d_out, int out_size, void* d_ws, size_t ws_size,
                              hipStream_t stream) {
    const float* xb  = (const float*)d_in[0];
    const float* Wq  = (const float*)d_in[1];
    const float* Wk  = (const float*)d_in[2];
    const float* Wv  = (const float*)d_in[3];
    const float* Wo  = (const float*)d_in[4];
    const float* pos = (const float*)d_in[5];
    float*       out = (float*)d_out;

    // Workspace carve-out (~136 MiB total)
    char*  ws  = (char*)d_ws;
    size_t off = 0;
    bf16* tokb = (bf16*)(ws + off); off += (size_t)M_ * C_ * sizeof(bf16);
    bf16* WqT  = (bf16*)(ws + off); off += (size_t)C_ * D_ * sizeof(bf16);
    bf16* WkT  = (bf16*)(ws + off); off += (size_t)C_ * D_ * sizeof(bf16);
    bf16* WvT  = (bf16*)(ws + off); off += (size_t)C_ * D_ * sizeof(bf16);
    bf16* WoT  = (bf16*)(ws + off); off += (size_t)D_ * C_ * sizeof(bf16);
    float* Qf  = (float*)(ws + off); off += (size_t)M_ * D_ * sizeof(float);
    bf16* Kb   = (bf16*)(ws + off); off += (size_t)M_ * D_ * sizeof(bf16);
    bf16* Vb   = (bf16*)(ws + off); off += (size_t)M_ * D_ * sizeof(bf16);
    bf16* ctxb = (bf16*)(ws + off); off += (size_t)M_ * D_ * sizeof(bf16);
    float* otok = (float*)(ws + off); off += (size_t)M_ * C_ * sizeof(float);
    (void)ws_size; (void)in_sizes; (void)n_in; (void)out_size;

    // 1) tokens = mean over T (bf16), NT reads of xb
    mean_kernel<<<(M_ * C_) / 256, 256, 0, stream>>>((const v4f*)xb, tokb);

    // 2) weight transpose+convert (N-major bf16 for WMMA B fragments)
    int wblocks = (C_ * D_ + 255) / 256;
    wtrans_kernel<<<wblocks, 256, 0, stream>>>(Wq, WqT, C_, D_);
    wtrans_kernel<<<wblocks, 256, 0, stream>>>(Wk, WkT, C_, D_);
    wtrans_kernel<<<wblocks, 256, 0, stream>>>(Wv, WvT, C_, D_);
    wtrans_kernel<<<wblocks, 256, 0, stream>>>(Wo, WoT, D_, C_);

    // 3) Q/K/V projections (+pos). 64x64 tile per wave, 4 waves/block.
    int gemm_blocks = (M_ / 64) * (D_ / 64) / 4;   // 2048 tiles / 4 = 512
    gemm64_kernel<false, true><<<gemm_blocks, 128, 0, stream>>>(tokb, WqT, pos, Qf, nullptr, M_, D_, C_);
    gemm64_kernel<true,  true><<<gemm_blocks, 128, 0, stream>>>(tokb, WkT, pos, nullptr, Kb, M_, D_, C_);
    gemm64_kernel<true,  true><<<gemm_blocks, 128, 0, stream>>>(tokb, WvT, pos, nullptr, Vb, M_, D_, C_);

    // 4) local window attention -> ctx (bf16)
    attn_kernel<<<M_, 256, 0, stream>>>((const v4f*)Qf, (const v4bf*)Kb,
                                        (const v4bf*)Vb, (v4bf*)ctxb);

    // 5) out_tok = ctx @ Wo (f32)
    gemm64_kernel<false, false><<<gemm_blocks, 128, 0, stream>>>(ctxb, WoT, nullptr, otok, nullptr, M_, C_, D_);

    // 6) out = xb + out_tok broadcast over T (NT streams)
    addbcast_kernel<<<(M_ * C_) / 256, 256, 0, stream>>>((const v4f*)xb, otok,
                                                         (v4f*)out);
}
